// BCOP_90159953477699
// MI455X (gfx1250) — matrix operand order repaired and checked by reading, not compiled, not run on hardware
//
#include <hip/hip_runtime.h>

// ---------------------------------------------------------------------------
// BCOP conv for MI455X (gfx1250, wave32, WMMA, async global->LDS).
//
// Phase A (tiny, fp32): power-iteration sigma + scale, 20 fused Bjorck
//   iterations, PQ projections, block-orth, 3x3 matrix-conv, H-multiply,
//   pack weight as f16 hi/lo split, layout [co][k=(kh*3+kw)*256+ci].
// Phase B (154.6 GFLOP, compute bound): split x once into channel-last f16
//   hi/lo, then implicit-GEMM conv with v_wmma_f32_16x16x32_f16, 3-term
//   f16-split (hi*hi + hi*lo + lo*hi) for near-fp32 accuracy. B tiles are
//   staged with GLOBAL_LOAD_ASYNC_TO_LDS_B128 (double-buffered), A tiles via
//   b128 register prefetch, so copies overlap the WMMA stream.
// ---------------------------------------------------------------------------

typedef _Float16 v8h  __attribute__((ext_vector_type(8)));
typedef _Float16 v16h __attribute__((ext_vector_type(16)));
typedef float    v8f  __attribute__((ext_vector_type(8)));

#define C256 256
#define MATF 65536           // elements in a 256x256 matrix
#define KTOT 2304            // 9 * 256
#define XHALFS 33554432ull   // 32*256*64*64

#if defined(__gfx1250__) && __has_builtin(__builtin_amdgcn_global_load_async_to_lds_b128)
#define HAVE_ASYNC 1
#else
#define HAVE_ASYNC 0
#endif

#if HAVE_ASYNC
// builtin signature (from compiler diagnostic): (int4 AS1*, int4 AS3*, Ii, Ii)
typedef int v4i_vs __attribute__((vector_size(16)));
typedef __attribute__((address_space(1))) v4i_vs glob_v4i;
typedef __attribute__((address_space(3))) v4i_vs lds_v4i;
// copy 32 bytes (16 halfs) global -> LDS, per lane, tracked by ASYNCcnt
#define ASYNC_CP32(gp, lp)                                                     \
    do {                                                                       \
        __builtin_amdgcn_global_load_async_to_lds_b128(                        \
            (glob_v4i*)(gp), (lds_v4i*)(lp), 0, 0);                            \
        __builtin_amdgcn_global_load_async_to_lds_b128(                        \
            (glob_v4i*)(gp), (lds_v4i*)(lp), 16, 0);                           \
    } while (0)
#endif

__device__ inline void wait_async0()
{
#if HAVE_ASYNC
#if __has_builtin(__builtin_amdgcn_s_wait_asynccnt)
    __builtin_amdgcn_s_wait_asynccnt(0);
#else
    asm volatile("s_wait_asynccnt 0x0" ::: "memory");
#endif
#endif
}

// ---------------------------------------------------------------------------
// Generic 32x32-tile fp32 GEMM accumulator for the small weight-construction
// matmuls. 256 threads as 16x16, each owning a 2x2 micro-tile.
// ---------------------------------------------------------------------------
template <typename FA, typename FB>
__device__ inline void gemm_acc_32x32(float (&acc)[2][2], FA lda, FB ldb, int K,
                                      float* As, float* Bs)
{
    const int t  = threadIdx.x;
    const int tx = t & 15, ty = t >> 4;
    for (int k0 = 0; k0 < K; k0 += 32) {
        __syncthreads();
        for (int i = t; i < 1024; i += 256) {
            int r = i >> 5, c = i & 31;
            As[r * 33 + c] = lda(r, k0 + c);   // A[i0+r][k0+c]
            Bs[r * 33 + c] = ldb(k0 + r, c);   // B[k0+r][j0+c]
        }
        __syncthreads();
#pragma unroll 8
        for (int kk = 0; kk < 32; ++kk) {
            float a0 = As[ty * 33 + kk];
            float a1 = As[(ty + 16) * 33 + kk];
            float b0 = Bs[kk * 33 + tx];
            float b1 = Bs[kk * 33 + tx + 16];
            acc[0][0] += a0 * b0; acc[0][1] += a0 * b1;
            acc[1][0] += a1 * b0; acc[1][1] += a1 * b1;
        }
    }
}

__device__ inline float block_reduce_sum(float x, float* red, int t)
{
    red[t] = x; __syncthreads();
    for (int off = 128; off > 0; off >>= 1) {
        if (t < off) red[t] += red[t + off];
        __syncthreads();
    }
    float r = red[0];
    __syncthreads();
    return r;
}

// ---------------------------------------------------------------------------
// 1) Power iteration (10 iters) -> sigma, then w0 = A / sigma.  5 blocks.
// ---------------------------------------------------------------------------
__global__ __launch_bounds__(256) void power_scale_kernel(
    const float* __restrict__ A5, const float* __restrict__ u0,
    float* __restrict__ w0)
{
    __shared__ float u[C256], v[C256], red[C256];
    const int z = blockIdx.x, t = threadIdx.x;
    const float* A = A5 + (size_t)z * MATF;
    u[t] = u0[z * C256 + t];
    __syncthreads();
    float s = 1.0f;
    for (int it = 0; it < 10; ++it) {
        float a = 0.f;                       // v = A^T u
        for (int i = 0; i < C256; ++i) a += A[i * C256 + t] * u[i];
        float nv = sqrtf(block_reduce_sum(a * a, red, t));
        v[t] = a / nv;
        __syncthreads();
        float b = 0.f;                       // u = A v
        for (int j = 0; j < C256; ++j) b += A[t * C256 + j] * v[j];
        float nu = sqrtf(block_reduce_sum(b * b, red, t));
        u[t] = b / nu;
        __syncthreads();
        s = nu;                              // final sigma = ||A v_last||
    }
    const float inv = 1.0f / s;
    for (int idx = t; idx < MATF; idx += 256)
        w0[(size_t)z * MATF + idx] = A[idx] * inv;
}

// ---------------------------------------------------------------------------
// 2) One fused Bjorck iteration: wout = 1.5 w - 0.5 w (w^T w), per 32-column
//    slab. grid = (8 slabs, 1, 5 matrices).
// ---------------------------------------------------------------------------
__global__ __launch_bounds__(256) void bjorck_kernel(
    const float* __restrict__ w, float* __restrict__ wout)
{
    __shared__ float As[32 * 33], Bs[32 * 33];
    __shared__ float T[C256 * 33];           // (w^T w)[:, slab]
    const int z = blockIdx.z, j0 = blockIdx.x * 32;
    const float* W  = w    + (size_t)z * MATF;
    float*       WO = wout + (size_t)z * MATF;
    const int tx = threadIdx.x & 15, ty = threadIdx.x >> 4;

    for (int rt = 0; rt < 8; ++rt) {         // T[r0..r0+31][jc] = (w^T w)
        const int r0 = rt * 32;
        float acc[2][2] = {{0.f, 0.f}, {0.f, 0.f}};
        gemm_acc_32x32(acc,
            [&](int r, int k) { return W[k * C256 + (r0 + r)]; },
            [&](int k, int c) { return W[k * C256 + (j0 + c)]; },
            C256, As, Bs);
        __syncthreads();
        T[(r0 + ty) * 33 + tx]           = acc[0][0];
        T[(r0 + ty) * 33 + tx + 16]      = acc[0][1];
        T[(r0 + ty + 16) * 33 + tx]      = acc[1][0];
        T[(r0 + ty + 16) * 33 + tx + 16] = acc[1][1];
    }
    __syncthreads();
    for (int rt = 0; rt < 8; ++rt) {         // wout = 1.5 w - 0.5 w T
        const int r0 = rt * 32;
        float acc[2][2] = {{0.f, 0.f}, {0.f, 0.f}};
        gemm_acc_32x32(acc,
            [&](int r, int k) { return W[(r0 + r) * C256 + k]; },
            [&](int k, int c) { return T[k * 33 + c]; },
            C256, As, Bs);
        const int r = r0 + ty, c = j0 + tx;
        WO[r * C256 + c]             = 1.5f * W[r * C256 + c]             - 0.5f * acc[0][0];
        WO[r * C256 + c + 16]        = 1.5f * W[r * C256 + c + 16]        - 0.5f * acc[0][1];
        WO[(r + 16) * C256 + c]      = 1.5f * W[(r + 16) * C256 + c]      - 0.5f * acc[1][0];
        WO[(r + 16) * C256 + c + 16] = 1.5f * W[(r + 16) * C256 + c + 16] - 0.5f * acc[1][1];
        __syncthreads();
    }
}

// ---------------------------------------------------------------------------
// 3) PQ[z] = B B^T, B = ortho[z+1][:, :128].  grid = (8,8,4).
// ---------------------------------------------------------------------------
__global__ __launch_bounds__(256) void pq_kernel(
    const float* __restrict__ ortho, float* __restrict__ PQ)
{
    __shared__ float As[32 * 33], Bs[32 * 33];
    const int z = blockIdx.z, i0 = blockIdx.y * 32, j0 = blockIdx.x * 32;
    const float* O = ortho + (size_t)(z + 1) * MATF;
    const int tx = threadIdx.x & 15, ty = threadIdx.x >> 4;
    float acc[2][2] = {{0.f, 0.f}, {0.f, 0.f}};
    gemm_acc_32x32(acc,
        [&](int r, int k) { return O[(i0 + r) * C256 + k]; },
        [&](int k, int c) { return O[(j0 + c) * C256 + k]; },
        128, As, Bs);
    float* out = PQ + (size_t)z * MATF;
    out[(i0 + ty) * C256 + j0 + tx]             = acc[0][0];
    out[(i0 + ty) * C256 + j0 + tx + 16]        = acc[0][1];
    out[(i0 + ty + 16) * C256 + j0 + tx]        = acc[1][0];
    out[(i0 + ty + 16) * C256 + j0 + tx + 16]   = acc[1][1];
}

// ---------------------------------------------------------------------------
// 4) block_orth entries: out[z] = opA(P_{2g}) @ opB(P_{2g+1}).  grid=(8,8,8).
// ---------------------------------------------------------------------------
__global__ __launch_bounds__(256) void borth_kernel(
    const float* __restrict__ PQ, float* __restrict__ Pout)
{
    __shared__ float As[32 * 33], Bs[32 * 33];
    const int z = blockIdx.z, i0 = blockIdx.y * 32, j0 = blockIdx.x * 32;
    const int grp = z >> 2, a = (z >> 1) & 1, b = z & 1;
    const float* PA = PQ + (size_t)(2 * grp) * MATF;
    const float* PB = PQ + (size_t)(2 * grp + 1) * MATF;
    const int tx = threadIdx.x & 15, ty = threadIdx.x >> 4;
    float acc[2][2] = {{0.f, 0.f}, {0.f, 0.f}};
    gemm_acc_32x32(acc,
        [&](int r, int k) {
            float v = PA[(i0 + r) * C256 + k];
            return a ? (((i0 + r) == k ? 1.0f : 0.0f) - v) : v;
        },
        [&](int k, int c) {
            float v = PB[k * C256 + (j0 + c)];
            return b ? ((k == (j0 + c) ? 1.0f : 0.0f) - v) : v;
        },
        C256, As, Bs);
    float* out = Pout + (size_t)z * MATF;
    out[(i0 + ty) * C256 + j0 + tx]           = acc[0][0];
    out[(i0 + ty) * C256 + j0 + tx + 16]      = acc[0][1];
    out[(i0 + ty + 16) * C256 + j0 + tx]      = acc[1][0];
    out[(i0 + ty + 16) * C256 + j0 + tx + 16] = acc[1][1];
}

// ---------------------------------------------------------------------------
// 5) 2x2 (*) 2x2 matrix-conv -> 3x3:  r[i,j] = sum p[a,b] @ q[i-a,j-b].
// ---------------------------------------------------------------------------
__global__ __launch_bounds__(256) void rconv_kernel(
    const float* __restrict__ Pbuf, float* __restrict__ Rbuf)
{
    __shared__ float As[32 * 33], Bs[32 * 33];
    const int z = blockIdx.z, i0 = blockIdx.y * 32, j0 = blockIdx.x * 32;
    const int i = z / 3, j = z % 3;
    const int tx = threadIdx.x & 15, ty = threadIdx.x >> 4;
    float acc[2][2] = {{0.f, 0.f}, {0.f, 0.f}};
    for (int a = 0; a < 2; ++a)
        for (int b = 0; b < 2; ++b) {
            const int c = i - a, d = j - b;
            if (c < 0 || c > 1 || d < 0 || d > 1) continue;
            const float* P = Pbuf + (size_t)(a * 2 + b) * MATF;
            const float* Q = Pbuf + (size_t)(4 + c * 2 + d) * MATF;
            gemm_acc_32x32(acc,
                [&](int r, int k) { return P[(i0 + r) * C256 + k]; },
                [&](int k, int cc) { return Q[k * C256 + (j0 + cc)]; },
                C256, As, Bs);
        }
    float* out = Rbuf + (size_t)z * MATF;
    out[(i0 + ty) * C256 + j0 + tx]           = acc[0][0];
    out[(i0 + ty) * C256 + j0 + tx + 16]      = acc[0][1];
    out[(i0 + ty + 16) * C256 + j0 + tx]      = acc[1][0];
    out[(i0 + ty + 16) * C256 + j0 + tx + 16] = acc[1][1];
}

// ---------------------------------------------------------------------------
// 6) t = H @ r[i,j], pack f16 hi/lo into [co][(kh*3+kw)*256+ci].  grid=(8,8,9)
// ---------------------------------------------------------------------------
__global__ __launch_bounds__(256) void packH_kernel(
    const float* __restrict__ H, const float* __restrict__ Rbuf,
    _Float16* __restrict__ Wh, _Float16* __restrict__ Wl)
{
    __shared__ float As[32 * 33], Bs[32 * 33];
    const int z = blockIdx.z, i0 = blockIdx.y * 32, j0 = blockIdx.x * 32;
    const int i = z / 3, j = z % 3;          // i = kw, j = kh
    const float* R = Rbuf + (size_t)z * MATF;
    const int tx = threadIdx.x & 15, ty = threadIdx.x >> 4;
    float acc[2][2] = {{0.f, 0.f}, {0.f, 0.f}};
    gemm_acc_32x32(acc,
        [&](int r, int k) { return H[(i0 + r) * C256 + k]; },
        [&](int k, int c) { return R[k * C256 + (j0 + c)]; },
        C256, As, Bs);
    const int kbase = (j * 3 + i) * C256;    // (kh*3+kw)*256
#pragma unroll
    for (int rr = 0; rr < 2; ++rr)
#pragma unroll
        for (int cc = 0; cc < 2; ++cc) {
            const int ci = i0 + ty + rr * 16;     // row of t = ci
            const int co = j0 + tx + cc * 16;     // col of t = co
            const float v = acc[rr][cc];
            const _Float16 hi = (_Float16)v;
            const _Float16 lo = (_Float16)(v - (float)hi);
            Wh[(size_t)co * KTOT + kbase + ci] = hi;
            Wl[(size_t)co * KTOT + kbase + ci] = lo;
        }
}

// ---------------------------------------------------------------------------
// 7) Split x -> channel-last f16 hi/lo:  xcl[n][h][w][ci] (transpose in LDS).
//    grid = (8 ci-tiles, 2 w-tiles, 2048 n*h), 256 threads as 32x8.
// ---------------------------------------------------------------------------
__global__ __launch_bounds__(256) void splitx_kernel(
    const float* __restrict__ x, _Float16* __restrict__ xhi,
    _Float16* __restrict__ xlo)
{
    __shared__ float tile[32][33];
    const int ci0 = blockIdx.x * 32;
    const int w0  = blockIdx.y * 32;
    const int nh  = blockIdx.z;
    const int n = nh >> 6, h = nh & 63;
    const int tx = threadIdx.x & 31, ty = threadIdx.x >> 5;
#pragma unroll
    for (int r = 0; r < 4; ++r) {
        const int ci = ci0 + ty + r * 8;
        tile[ty + r * 8][tx] = x[(((size_t)n * C256 + ci) * 64 + h) * 64 + w0 + tx];
    }
    __syncthreads();
    const size_t obase = ((size_t)n * 64 + h) * 64;
#pragma unroll
    for (int r = 0; r < 4; ++r) {
        const int wl = ty + r * 8;
        const float v = tile[tx][wl];
        const _Float16 hi = (_Float16)v;
        const _Float16 lo = (_Float16)(v - (float)hi);
        const size_t o = (obase + w0 + wl) * C256 + ci0 + tx;
        xhi[o] = hi;
        xlo[o] = lo;
    }
}

// ---------------------------------------------------------------------------
// WMMA fragment loaders (documented 16-bit A/B VGPR layouts).
// ---------------------------------------------------------------------------
#define LDT 40   // LDS row stride in halfs (32 + 8 pad)

__device__ inline v16h ldfragA(const _Float16* p, int g)
{
    v8h lo = *(const v8h*)(p + 8 * g);
    v8h hi = *(const v8h*)(p + 16 + 8 * g);
    return __builtin_shufflevector(lo, hi, 0,1,2,3,4,5,6,7,8,9,10,11,12,13,14,15);
}
__device__ inline v16h ldfragB(const _Float16* p, int g)
{
    v8h lo = *(const v8h*)(p + 8 * g);
    v8h hi = *(const v8h*)(p + 16 + 8 * g);
    return __builtin_shufflevector(lo, hi, 0,1,2,3,4,5,6,7,8,9,10,11,12,13,14,15);
}

// ---------------------------------------------------------------------------
// 8) Fast conv: channel-last f16 hi/lo x, async-staged weights, pipelined.
//    M = 32*64*64, N = 256, K = 2304. Block tile 128x128, K-chunk 32,
//    8 waves (2 M-groups x 4 N-groups), wave tile 64x32 = 4x2 WMMA tiles.
//    LDS: A single stage (reg-prefetched), B double-buffered (async) = 60KB.
// ---------------------------------------------------------------------------
__global__ __launch_bounds__(256) void conv_wmma_cl_kernel(
    const _Float16* __restrict__ xhi, const _Float16* __restrict__ xlo,
    const _Float16* __restrict__ Wh, const _Float16* __restrict__ Wl,
    const float* __restrict__ bias, float* __restrict__ out)
{
    __shared__ __align__(16) _Float16 Ah[128 * LDT], Al[128 * LDT];
    __shared__ __align__(16) _Float16 Bh[2][128 * LDT], Bl[2][128 * LDT];

    const int t     = threadIdx.x;
    const int co0   = blockIdx.x * 128;
    const int mb    = blockIdx.y;
    const int n_img = mb >> 5;
    const int h0    = (mb & 31) * 2;

    // loader roles: 2 threads per row, 16 halfs each
    const int row2 = t >> 1;
    const int part = (t & 1) * 16;
    const int ahh  = row2 >> 6, aw = row2 & 63;
    const int hb   = h0 + ahh;

    const int wid = t >> 5, lane = t & 31;
    const int g = lane >> 4, li = lane & 15;
    const int wm = wid & 1, wn = wid >> 1;

    v8f acc[4][2];
    v8f vz = {};
#pragma unroll
    for (int mt = 0; mt < 4; ++mt)
#pragma unroll
        for (int nt = 0; nt < 2; ++nt) acc[mt][nt] = vz;

    const size_t xrow_n = (size_t)n_img * 4096;   // row index base (n*64*64)

    auto aoff = [&](int step) -> size_t {
        const int kp = step >> 3, cc = step & 7;
        const int kh = kp / 3, kw = kp - kh * 3;
        const int hs = (hb + kh + 63) & 63;
        const int wsrc = (aw + kw + 63) & 63;
        return ((xrow_n + (size_t)hs * 64 + wsrc) << 8) + (cc * 32 + part);
    };
    auto boff = [&](int step) -> size_t {
        const int kp = step >> 3, cc = step & 7;
        return (size_t)(co0 + row2) * KTOT + kp * 256 + cc * 32 + part;
    };

    uint4 pa0, pa1, pa2, pa3;                   // A prefetch regs
#if !HAVE_ASYNC
    uint4 pb0, pb1, pb2, pb3;                   // B prefetch regs (fallback)
#endif

    auto issueA = [&](int step) {
        const size_t o = aoff(step);
        const uint4* ph = (const uint4*)(xhi + o);
        const uint4* pl = (const uint4*)(xlo + o);
        pa0 = ph[0]; pa1 = ph[1]; pa2 = pl[0]; pa3 = pl[1];
    };
    auto storeA = [&]() {
        *(uint4*)&Ah[row2 * LDT + part]     = pa0;
        *(uint4*)&Ah[row2 * LDT + part + 8] = pa1;
        *(uint4*)&Al[row2 * LDT + part]     = pa2;
        *(uint4*)&Al[row2 * LDT + part + 8] = pa3;
    };
    auto issueB = [&](int step, int stage) {
        const size_t o = boff(step);
#if HAVE_ASYNC
        ASYNC_CP32(Wh + o, &Bh[stage][row2 * LDT + part]);
        ASYNC_CP32(Wl + o, &Bl[stage][row2 * LDT + part]);
#else
        const uint4* ph = (const uint4*)(Wh + o);
        const uint4* pl = (const uint4*)(Wl + o);
        pb0 = ph[0]; pb1 = ph[1]; pb2 = pl[0]; pb3 = pl[1];
        (void)stage;
#endif
    };
#if !HAVE_ASYNC
    auto storeB = [&](int stage) {
        *(uint4*)&Bh[stage][row2 * LDT + part]     = pb0;
        *(uint4*)&Bh[stage][row2 * LDT + part + 8] = pb1;
        *(uint4*)&Bl[stage][row2 * LDT + part]     = pb2;
        *(uint4*)&Bl[stage][row2 * LDT + part + 8] = pb3;
    };
#endif

    // prologue: stage chunk 0
    issueA(0);
    issueB(0, 0);
    storeA();
#if !HAVE_ASYNC
    storeB(0);
#endif
    wait_async0();
    __syncthreads();

    for (int step = 0; step < 72; ++step) {
        const int cur = step & 1, nxt = cur ^ 1;
        // stage chunk step+1 (phantom final prefetch stays in-bounds of ws)
        issueA(step + 1);
        issueB(step + 1, nxt);

        // ---- compute from Ah/Al and Bh/Bl[cur] ----
        v16h bhv[2], blv[2];
#pragma unroll
        for (int nt = 0; nt < 2; ++nt) {
            const int nr = wn * 32 + nt * 16 + li;
            bhv[nt] = ldfragB(&Bh[cur][nr * LDT], g);
            blv[nt] = ldfragB(&Bl[cur][nr * LDT], g);
        }
#pragma unroll
        for (int mt = 0; mt < 4; ++mt) {
            const int mr = wm * 64 + mt * 16 + li;
            v16h ahv = ldfragA(&Ah[mr * LDT], g);
            v16h alv = ldfragA(&Al[mr * LDT], g);
#pragma unroll
            for (int nt = 0; nt < 2; ++nt) {
                acc[mt][nt] = __builtin_amdgcn_wmma_f32_16x16x32_f16(
                    false, ahv, false, bhv[nt], (short)0, acc[mt][nt], false, false);
                acc[mt][nt] = __builtin_amdgcn_wmma_f32_16x16x32_f16(
                    false, ahv, false, blv[nt], (short)0, acc[mt][nt], false, false);
                acc[mt][nt] = __builtin_amdgcn_wmma_f32_16x16x32_f16(
                    false, alv, false, bhv[nt], (short)0, acc[mt][nt], false, false);
            }
        }
        __syncthreads();       // all reads of Ah/Al (and Bh/Bl[cur]) done
        storeA();
#if !HAVE_ASYNC
        storeB(nxt);
#endif
        wait_async0();
        __syncthreads();       // chunk step+1 fully staged
    }

    // ---- epilogue: lane holds 8 consecutive w values (C layout M = 8g+r) ----
#pragma unroll
    for (int nt = 0; nt < 2; ++nt) {
        const int co = co0 + wn * 32 + nt * 16 + li;
        const float bv = bias[co];
#pragma unroll
        for (int mt = 0; mt < 4; ++mt) {
            const int ml = wm * 64 + mt * 16 + 8 * g;
            const int hh = ml >> 6, w0 = ml & 63;
            const size_t o =
                ((size_t)(n_img * C256 + co) * 64 + (h0 + hh)) * 64 + w0;
            float4 v0 = make_float4(acc[mt][nt][0] + bv, acc[mt][nt][1] + bv,
                                    acc[mt][nt][2] + bv, acc[mt][nt][3] + bv);
            float4 v1 = make_float4(acc[mt][nt][4] + bv, acc[mt][nt][5] + bv,
                                    acc[mt][nt][6] + bv, acc[mt][nt][7] + bv);
            *(float4*)(out + o)     = v0;
            *(float4*)(out + o + 4) = v1;
        }
    }
}

// ---------------------------------------------------------------------------
// 9) Fallback conv (direct fp32 x, split on the fly) for small workspaces.
// ---------------------------------------------------------------------------
__global__ __launch_bounds__(256) void conv_wmma_kernel(
    const float* __restrict__ x, const _Float16* __restrict__ Wh,
    const _Float16* __restrict__ Wl, const float* __restrict__ bias,
    float* __restrict__ out)
{
    __shared__ __align__(16) _Float16 Ah[128 * LDT], Al[128 * LDT];
    __shared__ __align__(16) _Float16 Bh[128 * LDT], Bl[128 * LDT];

    const int t     = threadIdx.x;
    const int co0   = blockIdx.x * 128;
    const int mb    = blockIdx.y;
    const int n_img = mb >> 5;
    const int h0    = (mb & 31) * 2;

    const int am  = t & 127;
    const int akc = (t >> 7) * 16;
    const int ahh = am >> 6;
    const int aw  = am & 63;
    const int ah_ = h0 + ahh;

    const int brow = t >> 1;
    const int bcol = (t & 1) * 16;

    const int wid = t >> 5, lane = t & 31;
    const int g = lane >> 4, li = lane & 15;
    const int wm = wid & 1, wn = wid >> 1;

    v8f acc[4][2];
    v8f vz = {};
#pragma unroll
    for (int mt = 0; mt < 4; ++mt)
#pragma unroll
        for (int nt = 0; nt < 2; ++nt) acc[mt][nt] = vz;

    const size_t xbase = (size_t)n_img * C256 * 4096;

    for (int kp = 0; kp < 9; ++kp) {
        const int kh = kp / 3, kw = kp % 3;
        const int hs = (ah_ + kh + 63) & 63;
        const int wsrc = (aw + kw + 63) & 63;
        const float* xp = x + xbase + (size_t)hs * 64 + wsrc;
        const _Float16* whp = Wh + (size_t)(co0 + brow) * KTOT + kp * C256 + bcol;
        const _Float16* wlp = Wl + (size_t)(co0 + brow) * KTOT + kp * C256 + bcol;

        for (int cc = 0; cc < 8; ++cc) {
            const int ci0 = cc * 32;
            __syncthreads();
#pragma unroll
            for (int e = 0; e < 16; ++e) {
                float v = xp[(size_t)(ci0 + akc + e) * 4096];
                _Float16 hi = (_Float16)v;
                _Float16 lo = (_Float16)(v - (float)hi);
                Ah[am * LDT + akc + e] = hi;
                Al[am * LDT + akc + e] = lo;
            }
            {
                const uint4* sh = (const uint4*)(whp + ci0);
                const uint4* sl = (const uint4*)(wlp + ci0);
                uint4 h0v = sh[0], h1v = sh[1];
                uint4 l0v = sl[0], l1v = sl[1];
                *(uint4*)(&Bh[brow * LDT + bcol])     = h0v;
                *(uint4*)(&Bh[brow * LDT + bcol + 8]) = h1v;
                *(uint4*)(&Bl[brow * LDT + bcol])     = l0v;
                *(uint4*)(&Bl[brow * LDT + bcol + 8]) = l1v;
            }
            __syncthreads();
            v16h bhv[2], blv[2];
#pragma unroll
            for (int nt = 0; nt < 2; ++nt) {
                const int nr = wn * 32 + nt * 16 + li;
                bhv[nt] = ldfragB(&Bh[nr * LDT], g);
                blv[nt] = ldfragB(&Bl[nr * LDT], g);
            }
#pragma unroll
            for (int mt = 0; mt < 4; ++mt) {
                const int mr = wm * 64 + mt * 16 + li;
                v16h ahv = ldfragA(&Ah[mr * LDT], g);
                v16h alv = ldfragA(&Al[mr * LDT], g);
#pragma unroll
                for (int nt = 0; nt < 2; ++nt) {
                    acc[mt][nt] = __builtin_amdgcn_wmma_f32_16x16x32_f16(
                        false, ahv, false, bhv[nt], (short)0, acc[mt][nt], false, false);
                    acc[mt][nt] = __builtin_amdgcn_wmma_f32_16x16x32_f16(
                        false, ahv, false, blv[nt], (short)0, acc[mt][nt], false, false);
                    acc[mt][nt] = __builtin_amdgcn_wmma_f32_16x16x32_f16(
                        false, alv, false, bhv[nt], (short)0, acc[mt][nt], false, false);
                }
            }
        }
    }

#pragma unroll
    for (int nt = 0; nt < 2; ++nt) {
        const int co = co0 + wn * 32 + nt * 16 + li;
        const float bv = bias[co];
#pragma unroll
        for (int mt = 0; mt < 4; ++mt) {
            const int ml = wm * 64 + mt * 16 + 8 * g;
            const int hh = ml >> 6, w0 = ml & 63;
            const size_t o =
                ((size_t)(n_img * C256 + co) * 64 + (h0 + hh)) * 64 + w0;
            float4 v0 = make_float4(acc[mt][nt][0] + bv, acc[mt][nt][1] + bv,
                                    acc[mt][nt][2] + bv, acc[mt][nt][3] + bv);
            float4 v1 = make_float4(acc[mt][nt][4] + bv, acc[mt][nt][5] + bv,
                                    acc[mt][nt][6] + bv, acc[mt][nt][7] + bv);
            *(float4*)(out + o)     = v0;
            *(float4*)(out + o + 4) = v1;
        }
    }
}

// ---------------------------------------------------------------------------
extern "C" void kernel_launch(void* const* d_in, const int* in_sizes, int n_in,
                              void* d_out, int out_size, void* d_ws, size_t ws_size,
                              hipStream_t stream)
{
    (void)in_sizes; (void)n_in; (void)out_size;
    const float* x     = (const float*)d_in[0];   // [32,256,64,64]
    const float* param = (const float*)d_in[1];   // [5,256,256]
    const float* u0    = (const float*)d_in[2];   // [5,256,1]
    const float* bias  = (const float*)d_in[3];   // [256]
    float* out = (float*)d_out;

    // workspace layout
    float* bufA = (float*)d_ws;                   // 5 matrices
    float* bufB = bufA + 5 * MATF;                // 5 matrices
    float* PQb  = bufB + 5 * MATF;                // 4 matrices
    float* Pb   = PQb + 4 * MATF;                 // 8 matrices (p0..3, q0..3)
    float* Rb   = Pb + 8 * MATF;                  // 9 matrices
    _Float16* Whp = (_Float16*)(Rb + 9 * MATF);   // 256*2304 f16 (hi)
    _Float16* Wlp = Whp + (size_t)C256 * KTOT;    // 256*2304 f16 (lo)
    _Float16* xhi = Wlp + (size_t)C256 * KTOT;    // 32*256*64*64 f16 (hi)
    _Float16* xlo = xhi + XHALFS;                 // 32*256*64*64 f16 (lo)
    const size_t need_fast =
        (size_t)((char*)(xlo + XHALFS) - (char*)d_ws);  // ~145 MB

    // Phase A: weight construction
    power_scale_kernel<<<dim3(5), 256, 0, stream>>>(param, u0, bufA);
    float* cur = bufA; float* nxt = bufB;
    for (int it = 0; it < 20; ++it) {
        bjorck_kernel<<<dim3(8, 1, 5), 256, 0, stream>>>(cur, nxt);
        float* tmp = cur; cur = nxt; nxt = tmp;
    }
    pq_kernel   <<<dim3(8, 8, 4), 256, 0, stream>>>(cur, PQb);
    borth_kernel<<<dim3(8, 8, 8), 256, 0, stream>>>(PQb, Pb);
    rconv_kernel<<<dim3(8, 8, 9), 256, 0, stream>>>(Pb, Rb);
    packH_kernel<<<dim3(8, 8, 9), 256, 0, stream>>>(cur /*H*/, Rb, Whp, Wlp);

    // Phase B
    if (ws_size >= need_fast) {
        splitx_kernel<<<dim3(8, 2, 2048), 256, 0, stream>>>(x, xhi, xlo);
        conv_wmma_cl_kernel<<<dim3(2, 1024), 256, 0, stream>>>(
            xhi, xlo, Whp, Wlp, bias, out);
    } else {
        conv_wmma_kernel<<<dim3(2, 1024), 256, 0, stream>>>(
            x, Whp, Wlp, bias, out);
    }
}
